// FusedMoE_5437428596962
// MI455X (gfx1250) — matrix operand order repaired and checked by reading, not compiled, not run on hardware
//
#include <hip/hip_runtime.h>

// ---------------- problem constants (from reference) ----------------
#define T_TOK 8192
#define H_DIM 1024
#define I_DIM 2816
#define NE    8
#define TOPK  2

// ---------------- GEMM tiling ----------------
#define BM  128
#define BN  64
#define BK  32
#define BKP 40   // padded LDS row pitch in bf16 elems (80B, 16B-aligned, destaggers banks)

typedef __attribute__((ext_vector_type(16))) __bf16 v16bf;
typedef __attribute__((ext_vector_type(8)))  float  v8f;
typedef __attribute__((ext_vector_type(4)))  int    v4i_a;

// ---------------- CDNA5 async global->LDS copy (16B per lane) ----------------
#if __has_builtin(__builtin_amdgcn_global_load_async_to_lds_b128)
__device__ __forceinline__ void async_cp16(const void* gp, void* lp) {
  __attribute__((address_space(1))) v4i_a* g =
      (__attribute__((address_space(1))) v4i_a*)(
          (__attribute__((address_space(1))) void*)const_cast<void*>(gp));
  __attribute__((address_space(3))) v4i_a* l =
      (__attribute__((address_space(3))) v4i_a*)(
          (__attribute__((address_space(3))) void*)lp);
  __builtin_amdgcn_global_load_async_to_lds_b128(g, l, 0, 0);
}
#define ASYNC_CP16(gp, lp) async_cp16((const void*)(gp), (void*)(lp))
#else
__device__ __forceinline__ unsigned __lds_u32(const void* p) {
  return (unsigned)(size_t)(const __attribute__((address_space(3))) void*)p;
}
#define ASYNC_CP16(gp, lp)                                                  \
  asm volatile("global_load_async_to_lds_b128 %0, %1, off"                  \
               :: "v"(__lds_u32(lp)), "v"(gp) : "memory")
#endif

#if __has_builtin(__builtin_amdgcn_s_wait_asynccnt)
#define WAIT_ASYNC() __builtin_amdgcn_s_wait_asynccnt(0)
#else
#define WAIT_ASYNC() asm volatile("s_wait_asynccnt 0x0" ::: "memory")
#endif

// ---------------- fp32 -> bf16 conversion ----------------
__global__ void conv_bf16_kernel(const float* __restrict__ in, __bf16* __restrict__ out,
                                 long long n) {
  long long i = (long long)blockIdx.x * blockDim.x + threadIdx.x;
  long long stride = (long long)gridDim.x * blockDim.x;
  for (; i < n; i += stride) out[i] = (__bf16)in[i];
}

// ---------------- router: softmax + top-2 + renorm ----------------
__global__ void router_topk_kernel(const float* __restrict__ logits,
                                   int* __restrict__ topk_id, float* __restrict__ topk_w,
                                   int* __restrict__ counts) {
  int t = blockIdx.x * blockDim.x + threadIdx.x;
  if (t >= T_TOK) return;
  float l[NE];
  float m = -1e30f;
  for (int e = 0; e < NE; ++e) { l[e] = logits[t * NE + e]; m = fmaxf(m, l[e]); }
  for (int e = 0; e < NE; ++e) l[e] = __expf(l[e] - m);  // softmax norm cancels in renorm
  int e0 = 0;
  for (int e = 1; e < NE; ++e) if (l[e] > l[e0]) e0 = e;
  int e1 = (e0 == 0) ? 1 : 0;
  for (int e = 0; e < NE; ++e) if (e != e0 && l[e] > l[e1]) e1 = e;
  float p0 = l[e0], p1 = l[e1];
  float inv = 1.0f / (p0 + p1);
  topk_id[t * 2 + 0] = e0; topk_w[t * 2 + 0] = p0 * inv;
  topk_id[t * 2 + 1] = e1; topk_w[t * 2 + 1] = p1 * inv;
  atomicAdd(&counts[e0], 1);
  atomicAdd(&counts[e1], 1);
}

// ---------------- tiny exclusive scan over 8 experts ----------------
__global__ void scan_offsets_kernel(const int* __restrict__ counts, int* __restrict__ offsets) {
  if (threadIdx.x == 0 && blockIdx.x == 0) {
    int o = 0;
    for (int e = 0; e < NE; ++e) { offsets[e] = o; o += counts[e]; }
  }
}

// ---------------- compact (token, weight) into per-expert slots ----------------
__global__ void assign_slots_kernel(const int* __restrict__ topk_id, const float* __restrict__ topk_w,
                                    const int* __restrict__ offsets, int* __restrict__ fill,
                                    int* __restrict__ tok_map, float* __restrict__ tok_w) {
  int t = blockIdx.x * blockDim.x + threadIdx.x;
  if (t >= T_TOK) return;
  #pragma unroll
  for (int k = 0; k < TOPK; ++k) {
    int e = topk_id[t * 2 + k];
    int pos = atomicAdd(&fill[e], 1);
    int slot = offsets[e] + pos;
    tok_map[slot] = t;
    tok_w[slot] = topk_w[t * 2 + k];
  }
}

// ---------------- stage A: act = silu(x@Wg^T) * (x@Wu^T) * route_w  (bf16 WMMA) ----------------
__launch_bounds__(256)
__global__ void moe_gateup_kernel(const __bf16* __restrict__ xb, const __bf16* __restrict__ w13b,
                                  __bf16* __restrict__ actb,
                                  const int* __restrict__ tok_map, const float* __restrict__ tok_w,
                                  const int* __restrict__ counts, const int* __restrict__ offsets) {
  __shared__ __bf16 As[2][BM * BKP];
  __shared__ __bf16 Bg[2][BN * BKP];
  __shared__ __bf16 Bu[2][BN * BKP];

  const int e = blockIdx.z;
  const int cnt = counts[e];
  const int rowBase = blockIdx.y * BM;
  if (rowBase >= cnt) return;
  const int v = (cnt - rowBase < BM) ? (cnt - rowBase) : BM;
  const int slotBase = offsets[e] + rowBase;
  const int nBase = blockIdx.x * BN;

  const int tid  = threadIdx.x;
  const int lane = tid & 31;
  const int wave = tid >> 5;
  const int wm = wave & 3;    // 0..3 -> 32-row group
  const int wn = wave >> 2;   // 0..1 -> 32-col group
  const int lr  = tid >> 2;        // load row 0..63
  const int seg = (tid & 3) * 8;   // bf16 element offset within row (16B chunks)

  // A-tile gather rows (clamp invalid rows to a valid slot; their outputs are discarded)
  const int r0 = lr, r1 = lr + 64;
  const int c0 = (r0 < v) ? r0 : (v - 1);
  const int c1 = (r1 < v) ? r1 : (v - 1);
  const size_t aBase0 = (size_t)tok_map[slotBase + c0] * H_DIM;
  const size_t aBase1 = (size_t)tok_map[slotBase + c1] * H_DIM;
  const size_t gRow   = ((size_t)e * 2 * I_DIM + (size_t)(nBase + lr)) * H_DIM;
  const size_t uRow   = gRow + (size_t)I_DIM * H_DIM;

  v8f g[2][2] = {}, u[2][2] = {};
  const int mrow = lane & 15;
  const int hi   = lane >> 4;

  auto issue = [&](int k0, int b) {
    ASYNC_CP16(xb + aBase0 + k0 + seg, &As[b][r0 * BKP + seg]);
    ASYNC_CP16(xb + aBase1 + k0 + seg, &As[b][r1 * BKP + seg]);
    ASYNC_CP16(w13b + gRow + k0 + seg, &Bg[b][lr * BKP + seg]);
    ASYNC_CP16(w13b + uRow + k0 + seg, &Bu[b][lr * BKP + seg]);
  };

  issue(0, 0);
  const int NSTEP = H_DIM / BK;
  for (int i = 0; i < NSTEP; ++i) {
    const int b = i & 1;
    WAIT_ASYNC();        // this wave's copies into buf b are done
    __syncthreads();     // everyone's copies are done; everyone left buf b^1
    if (i + 1 < NSTEP) issue((i + 1) * BK, b ^ 1);

    v16bf a0, a1, bg0, bg1, bu0, bu1;
    {
      const __bf16* pa0 = &As[b][(wm * 32 + mrow) * BKP];
      const __bf16* pa1 = pa0 + 16 * BKP;
      #pragma unroll
      for (int i2 = 0; i2 < 16; ++i2) {
        a0[i2] = pa0[hi * 16 + i2];
        a1[i2] = pa1[hi * 16 + i2];
      }
      const __bf16* pg0 = &Bg[b][(wn * 32 + mrow) * BKP + hi * 16];
      const __bf16* pg1 = pg0 + 16 * BKP;
      const __bf16* pu0 = &Bu[b][(wn * 32 + mrow) * BKP + hi * 16];
      const __bf16* pu1 = pu0 + 16 * BKP;
      #pragma unroll
      for (int i2 = 0; i2 < 16; ++i2) {
        bg0[i2] = pg0[i2]; bg1[i2] = pg1[i2];
        bu0[i2] = pu0[i2]; bu1[i2] = pu1[i2];
      }
    }
    g[0][0] = __builtin_amdgcn_wmma_f32_16x16x32_bf16(false, a0, false, bg0, (short)0, g[0][0], false, false);
    g[0][1] = __builtin_amdgcn_wmma_f32_16x16x32_bf16(false, a0, false, bg1, (short)0, g[0][1], false, false);
    g[1][0] = __builtin_amdgcn_wmma_f32_16x16x32_bf16(false, a1, false, bg0, (short)0, g[1][0], false, false);
    g[1][1] = __builtin_amdgcn_wmma_f32_16x16x32_bf16(false, a1, false, bg1, (short)0, g[1][1], false, false);
    u[0][0] = __builtin_amdgcn_wmma_f32_16x16x32_bf16(false, a0, false, bu0, (short)0, u[0][0], false, false);
    u[0][1] = __builtin_amdgcn_wmma_f32_16x16x32_bf16(false, a0, false, bu1, (short)0, u[0][1], false, false);
    u[1][0] = __builtin_amdgcn_wmma_f32_16x16x32_bf16(false, a1, false, bu0, (short)0, u[1][0], false, false);
    u[1][1] = __builtin_amdgcn_wmma_f32_16x16x32_bf16(false, a1, false, bu1, (short)0, u[1][1], false, false);
  }

  // epilogue: silu(gate)*up, scaled by routing weight (legal: down-proj is linear)
  const int colBase = nBase + wn * 32 + (lane & 15);
  #pragma unroll
  for (int m = 0; m < 2; ++m) {
    #pragma unroll
    for (int r = 0; r < 8; ++r) {
      int rl = wm * 32 + m * 16 + hi * 8 + r;
      if (rl < v) {
        float wt = tok_w[slotBase + rl];
        size_t rowOff = (size_t)(slotBase + rl) * I_DIM;
        #pragma unroll
        for (int n = 0; n < 2; ++n) {
          float gg = g[m][n][r], uu = u[m][n][r];
          float sil = gg / (1.0f + __expf(-gg));
          actb[rowOff + colBase + n * 16] = (__bf16)(sil * uu * wt);
        }
      }
    }
  }
}

// ---------------- stage B: out[token] += act @ W2^T  (bf16 WMMA, f32 atomics) ----------------
__launch_bounds__(256)
__global__ void moe_down_kernel(const __bf16* __restrict__ actb, const __bf16* __restrict__ w2b,
                                float* __restrict__ out,
                                const int* __restrict__ tok_map,
                                const int* __restrict__ counts, const int* __restrict__ offsets) {
  __shared__ __bf16 As[2][BM * BKP];
  __shared__ __bf16 Bs[2][BN * BKP];

  const int e = blockIdx.z;
  const int cnt = counts[e];
  const int rowBase = blockIdx.y * BM;
  if (rowBase >= cnt) return;
  const int v = (cnt - rowBase < BM) ? (cnt - rowBase) : BM;
  const int slotBase = offsets[e] + rowBase;
  const int nBase = blockIdx.x * BN;

  const int tid  = threadIdx.x;
  const int lane = tid & 31;
  const int wave = tid >> 5;
  const int wm = wave & 3;
  const int wn = wave >> 2;
  const int lr  = tid >> 2;
  const int seg = (tid & 3) * 8;

  const int r0 = lr, r1 = lr + 64;
  const int c0 = (r0 < v) ? r0 : (v - 1);
  const int c1 = (r1 < v) ? r1 : (v - 1);
  const size_t aRow0 = (size_t)(slotBase + c0) * I_DIM;
  const size_t aRow1 = (size_t)(slotBase + c1) * I_DIM;
  const size_t bRow  = ((size_t)e * H_DIM + (size_t)(nBase + lr)) * I_DIM;

  v8f acc[2][2] = {};
  const int mrow = lane & 15;
  const int hi   = lane >> 4;

  auto issue = [&](int k0, int b) {
    ASYNC_CP16(actb + aRow0 + k0 + seg, &As[b][r0 * BKP + seg]);
    ASYNC_CP16(actb + aRow1 + k0 + seg, &As[b][r1 * BKP + seg]);
    ASYNC_CP16(w2b + bRow + k0 + seg, &Bs[b][lr * BKP + seg]);
  };

  issue(0, 0);
  const int NSTEP = I_DIM / BK;
  for (int i = 0; i < NSTEP; ++i) {
    const int b = i & 1;
    WAIT_ASYNC();
    __syncthreads();
    if (i + 1 < NSTEP) issue((i + 1) * BK, b ^ 1);

    v16bf a0, a1, b0, b1;
    {
      const __bf16* pa0 = &As[b][(wm * 32 + mrow) * BKP];
      const __bf16* pa1 = pa0 + 16 * BKP;
      #pragma unroll
      for (int i2 = 0; i2 < 16; ++i2) {
        a0[i2] = pa0[hi * 16 + i2];
        a1[i2] = pa1[hi * 16 + i2];
      }
      const __bf16* pb0 = &Bs[b][(wn * 32 + mrow) * BKP + hi * 16];
      const __bf16* pb1 = pb0 + 16 * BKP;
      #pragma unroll
      for (int i2 = 0; i2 < 16; ++i2) { b0[i2] = pb0[i2]; b1[i2] = pb1[i2]; }
    }
    acc[0][0] = __builtin_amdgcn_wmma_f32_16x16x32_bf16(false, a0, false, b0, (short)0, acc[0][0], false, false);
    acc[0][1] = __builtin_amdgcn_wmma_f32_16x16x32_bf16(false, a0, false, b1, (short)0, acc[0][1], false, false);
    acc[1][0] = __builtin_amdgcn_wmma_f32_16x16x32_bf16(false, a1, false, b0, (short)0, acc[1][0], false, false);
    acc[1][1] = __builtin_amdgcn_wmma_f32_16x16x32_bf16(false, a1, false, b1, (short)0, acc[1][1], false, false);
  }

  const int colBase = nBase + wn * 32 + (lane & 15);
  #pragma unroll
  for (int m = 0; m < 2; ++m) {
    #pragma unroll
    for (int r = 0; r < 8; ++r) {
      int rl = wm * 32 + m * 16 + hi * 8 + r;
      if (rl < v) {
        int token = tok_map[slotBase + rl];
        size_t rowOff = (size_t)token * H_DIM;
        #pragma unroll
        for (int n = 0; n < 2; ++n) {
          atomicAdd(&out[rowOff + colBase + n * 16], acc[m][n][r]);
        }
      }
    }
  }
}

// ---------------- host launcher ----------------
extern "C" void kernel_launch(void* const* d_in, const int* in_sizes, int n_in,
                              void* d_out, int out_size, void* d_ws, size_t ws_size,
                              hipStream_t stream) {
  (void)in_sizes; (void)n_in; (void)out_size; (void)ws_size;

  const float* x      = (const float*)d_in[0];
  const float* logits = (const float*)d_in[1];
  const float* w13    = (const float*)d_in[2];
  const float* w2     = (const float*)d_in[3];
  // d_in[4] = top_k (fixed at 2 by the reference setup)
  float* out = (float*)d_out;

  char* ws = (char*)d_ws;
  size_t off = 0;
  auto carve = [&](size_t bytes) -> void* {
    void* p = ws + off;
    off = (off + bytes + 255) & ~(size_t)255;
    return p;
  };

  __bf16* xb    = (__bf16*)carve((size_t)T_TOK * H_DIM * 2);
  __bf16* w13b  = (__bf16*)carve((size_t)NE * 2 * I_DIM * H_DIM * 2);
  __bf16* w2b   = (__bf16*)carve((size_t)NE * H_DIM * I_DIM * 2);
  __bf16* actb  = (__bf16*)carve((size_t)T_TOK * TOPK * I_DIM * 2);
  int*    topk_id = (int*)carve((size_t)T_TOK * TOPK * 4);
  float*  topk_w  = (float*)carve((size_t)T_TOK * TOPK * 4);
  int*    tok_map = (int*)carve((size_t)T_TOK * TOPK * 4);
  float*  tok_w   = (float*)carve((size_t)T_TOK * TOPK * 4);
  int*    meta    = (int*)carve(3 * NE * 4);   // counts | offsets | fill
  int* counts  = meta;
  int* offsets = meta + NE;
  int* fill    = meta + 2 * NE;

  (void)hipMemsetAsync(out, 0, (size_t)T_TOK * H_DIM * 4, stream);
  (void)hipMemsetAsync(meta, 0, 3 * NE * 4, stream);

  // 1) precision: fp32 -> bf16
  conv_bf16_kernel<<<2048, 256, 0, stream>>>(x,   xb,   (long long)T_TOK * H_DIM);
  conv_bf16_kernel<<<2048, 256, 0, stream>>>(w13, w13b, (long long)NE * 2 * I_DIM * H_DIM);
  conv_bf16_kernel<<<2048, 256, 0, stream>>>(w2,  w2b,  (long long)NE * H_DIM * I_DIM);

  // 2) routing
  router_topk_kernel<<<T_TOK / 256, 256, 0, stream>>>(logits, topk_id, topk_w, counts);
  scan_offsets_kernel<<<1, 32, 0, stream>>>(counts, offsets);
  assign_slots_kernel<<<T_TOK / 256, 256, 0, stream>>>(topk_id, topk_w, offsets, fill,
                                                       tok_map, tok_w);

  // 3) grouped GEMMs (worst-case grid, blocks early-exit on counts[e])
  dim3 gridA(I_DIM / BN, T_TOK / BM, NE);   // 44 x 64 x 8
  moe_gateup_kernel<<<gridA, 256, 0, stream>>>(xb, w13b, actb, tok_map, tok_w, counts, offsets);

  dim3 gridB(H_DIM / BN, T_TOK / BM, NE);   // 16 x 64 x 8
  moe_down_kernel<<<gridB, 256, 0, stream>>>(actb, w2b, out, tok_map, counts, offsets);
}
